// Faithful_SAE_38826504356552
// MI455X (gfx1250) — matrix operand order repaired and checked by reading, not compiled, not run on hardware
//
#include <hip/hip_runtime.h>

// ---------------------------------------------------------------------------
// Faithful SAE forward, MI455X (gfx1250, wave32, WMMA):
//  1) latent = x @ enc : 550 GFLOP, compute-bound. bf16x3 split (hi+lo) WMMA
//     with f32 accumulators (~2^-16 rel. err), double-buffered LDS software
//     pipeline, NT stores for the 537MB latent stream (exceeds 192MB L2).
//  2) top-32 of 16384 per row: register-resident radix threshold search,
//     wave-reduced counts, deterministic index-ranked compaction.
//  3) recon = sparse @ dec as a 32-row gather per output row (decoder is
//     134MB -> L2-resident; ~1 GFLOP, bandwidth-trivial).
// ---------------------------------------------------------------------------

typedef __attribute__((ext_vector_type(16))) __bf16 v16bf;
typedef __attribute__((ext_vector_type(8)))  __bf16 v8bf;
typedef __attribute__((ext_vector_type(4)))  __bf16 v4bf;
typedef __attribute__((ext_vector_type(8)))  float  v8f;
typedef __attribute__((ext_vector_type(4)))  float  v4f;

#define INPUT_DIM  2048
#define LATENT_DIM 16384
#define HIDDEN_DIM 2048
#define TOPK       32

#define LDS_STRIDE 40              // 32 bf16 payload + 8 pad: conflict-free b128
#define PLANE      (128 * LDS_STRIDE)

// ======================= encode GEMM (bf16x3 split) ========================

__device__ __forceinline__ void stage_load(const float* __restrict__ X,
                                           const float* __restrict__ W,
                                           int m0, int n0, int k0, int t,
                                           v4f (&ra)[4], v4f (&rb)[4]) {
#pragma unroll
  for (int i = 0; i < 4; ++i) {
    const int idx = t + i * 256;
    ra[i] = *(const v4f*)(X + (size_t)(m0 + (idx >> 3)) * INPUT_DIM + k0 + (idx & 7) * 4);
    rb[i] = *(const v4f*)(W + (size_t)(k0 + (idx >> 5)) * LATENT_DIM + n0 + (idx & 31) * 4);
  }
}

__device__ __forceinline__ void stage_prefetch(const float* __restrict__ X,
                                               const float* __restrict__ W,
                                               int m0, int n0, int k0, int t) {
#pragma unroll
  for (int i = 0; i < 4; ++i) {
    const int idx = t + i * 256;
    __builtin_prefetch(X + (size_t)(m0 + (idx >> 3)) * INPUT_DIM + k0 + (idx & 7) * 4, 0, 1);
    __builtin_prefetch(W + (size_t)(k0 + (idx >> 5)) * LATENT_DIM + n0 + (idx & 31) * 4, 0, 1);
  }
}

__device__ __forceinline__ void stage_store(__bf16* __restrict__ Ahi, __bf16* __restrict__ Alo,
                                            __bf16* __restrict__ Bhi, __bf16* __restrict__ Blo,
                                            int t, const v4f (&ra)[4], const v4f (&rb)[4]) {
#pragma unroll
  for (int i = 0; i < 4; ++i) {
    const int idx = t + i * 256;
    {  // A: row-major [m][k], hi/lo split
      const int r = idx >> 3, c4 = idx & 7;
      v4bf h, l;
#pragma unroll
      for (int j = 0; j < 4; ++j) {
        __bf16 hb = (__bf16)ra[i][j];
        h[j] = hb;
        l[j] = (__bf16)(ra[i][j] - (float)hb);
      }
      *(v4bf*)&Ahi[r * LDS_STRIDE + c4 * 4] = h;
      *(v4bf*)&Alo[r * LDS_STRIDE + c4 * 4] = l;
    }
    {  // B: transposed into [n][k], hi/lo split
      const int k = idx >> 5, c4 = idx & 31;
#pragma unroll
      for (int j = 0; j < 4; ++j) {
        __bf16 hb = (__bf16)rb[i][j];
        Bhi[(c4 * 4 + j) * LDS_STRIDE + k] = hb;
        Blo[(c4 * 4 + j) * LDS_STRIDE + k] = (__bf16)(rb[i][j] - (float)hb);
      }
    }
  }
}

__device__ __forceinline__ void wave_compute(const __bf16* __restrict__ Ahi,
                                             const __bf16* __restrict__ Alo,
                                             const __bf16* __restrict__ Bhi,
                                             const __bf16* __restrict__ Blo,
                                             int wm, int wn, int lane, v8f (&acc)[4][2]) {
  const int mrow = lane & 15;
  const int kbA = (lane >> 4) * 8;    // A frag: lanes 0-15 K0..7/16..23, 16-31 K8..15/24..31
  const int kbB = (lane >> 4) * 8;    // B frag: lanes 0-15 K0..7/16..23, 16-31 K8..15/24..31

  v16bf bh[2], bl[2];
#pragma unroll
  for (int nt = 0; nt < 2; ++nt) {
    const int n = wn * 32 + nt * 16 + mrow;
    v8bf b0 = *(const v8bf*)&Bhi[n * LDS_STRIDE + kbB];
    v8bf b1 = *(const v8bf*)&Bhi[n * LDS_STRIDE + kbB + 16];
    bh[nt] = __builtin_shufflevector(b0, b1, 0,1,2,3,4,5,6,7,8,9,10,11,12,13,14,15);
    v8bf c0 = *(const v8bf*)&Blo[n * LDS_STRIDE + kbB];
    v8bf c1 = *(const v8bf*)&Blo[n * LDS_STRIDE + kbB + 16];
    bl[nt] = __builtin_shufflevector(c0, c1, 0,1,2,3,4,5,6,7,8,9,10,11,12,13,14,15);
  }
#pragma unroll
  for (int mt = 0; mt < 4; ++mt) {
    const int m = wm * 64 + mt * 16 + mrow;
    v8bf a0 = *(const v8bf*)&Ahi[m * LDS_STRIDE + kbA];
    v8bf a1 = *(const v8bf*)&Ahi[m * LDS_STRIDE + 16 + kbA];
    v16bf ah = __builtin_shufflevector(a0, a1, 0,1,2,3,4,5,6,7,8,9,10,11,12,13,14,15);
    v8bf d0 = *(const v8bf*)&Alo[m * LDS_STRIDE + kbA];
    v8bf d1 = *(const v8bf*)&Alo[m * LDS_STRIDE + 16 + kbA];
    v16bf al = __builtin_shufflevector(d0, d1, 0,1,2,3,4,5,6,7,8,9,10,11,12,13,14,15);
#pragma unroll
    for (int nt = 0; nt < 2; ++nt) {
      acc[mt][nt] = __builtin_amdgcn_wmma_f32_16x16x32_bf16(
          false, ah, false, bh[nt], (short)0, acc[mt][nt], false, false);
      acc[mt][nt] = __builtin_amdgcn_wmma_f32_16x16x32_bf16(
          false, ah, false, bl[nt], (short)0, acc[mt][nt], false, false);
      acc[mt][nt] = __builtin_amdgcn_wmma_f32_16x16x32_bf16(
          false, al, false, bh[nt], (short)0, acc[mt][nt], false, false);
    }
  }
}

__global__ __launch_bounds__(256) void sae_encode_bf16x3(
    const float* __restrict__ X,   // [M, 2048]
    const float* __restrict__ W,   // [2048, 16384]
    float* __restrict__ L)         // [M, 16384] (sparse region of d_out)
{
  // double-buffered: [buf][Ahi|Alo|Bhi|Blo]
  __shared__ __align__(16) __bf16 smem[2][4][PLANE];

  const int t    = threadIdx.x;
  const int lane = t & 31;
  const int wave = t >> 5;
  const int wm   = wave >> 2;          // 0..1 : 64-row slab
  const int wn   = wave & 3;           // 0..3 : 32-col slab
  const int m0   = blockIdx.y * 128;
  const int n0   = blockIdx.x * 128;

  v8f acc[4][2] = {};
  v4f ra[4], rb[4];

  // prologue: fill buffer 0
  stage_load(X, W, m0, n0, 0, t, ra, rb);
  stage_store(smem[0][0], smem[0][1], smem[0][2], smem[0][3], t, ra, rb);
  __syncthreads();

  const int NIT = INPUT_DIM / 32;
  for (int it = 0; it < NIT; ++it) {
    if (it + 1 < NIT) {
      stage_load(X, W, m0, n0, (it + 1) * 32, t, ra, rb);   // hide HBM latency
      if (it + 2 < NIT) stage_prefetch(X, W, m0, n0, (it + 2) * 32, t);
    }
    const int p = it & 1;
    wave_compute(smem[p][0], smem[p][1], smem[p][2], smem[p][3], wm, wn, lane, acc);
    if (it + 1 < NIT) {
      const int q = (it + 1) & 1;
      stage_store(smem[q][0], smem[q][1], smem[q][2], smem[q][3], t, ra, rb);
      __syncthreads();   // single barrier per iteration (double buffer)
    }
  }

  // epilogue: C layout = lanes 0-15 -> M+0..7, lanes 16-31 -> M+8..15.
  // Latent stream (537MB) > L2 (192MB): nontemporal stores.
#pragma unroll
  for (int mt = 0; mt < 4; ++mt)
#pragma unroll
    for (int nt = 0; nt < 2; ++nt) {
      const int col   = n0 + wn * 32 + nt * 16 + (lane & 15);
      const int rbase = m0 + wm * 64 + mt * 16 + (lane >> 4) * 8;
#pragma unroll
      for (int r = 0; r < 8; ++r)
        __builtin_nontemporal_store(acc[mt][nt][r],
                                    &L[(size_t)(rbase + r) * LATENT_DIM + col]);
    }
}

// ============================ top-k (K=32) =================================
// One WG per row; 64 order-preserving u32 keys per thread held in registers;
// 32-round radix binary search; wave shfl-reduce + 1 atomic per wave per
// round; deterministic compaction by index rank.
__global__ __launch_bounds__(256) void sae_topk(
    float* __restrict__ L,          // in: latent row, out: sparse row (in place)
    float* __restrict__ outv,       // [M, 32]
    int*   __restrict__ outi)       // [M, 32]
{
  __shared__ int   s_cnt;
  __shared__ int   s_sel;
  __shared__ float sv[TOPK];
  __shared__ int   si[TOPK];

  const int row = blockIdx.x;
  const int t   = threadIdx.x;
  float* Lrow = L + (size_t)row * LATENT_DIM;

  // element index for (chunk ch, slot e): ch*1024 + t*4 + e  (b128 coalesced)
  unsigned key[64];
#pragma unroll
  for (int ch = 0; ch < 16; ++ch) {
    v4f v = *(const v4f*)(Lrow + ch * 1024 + t * 4);
#pragma unroll
    for (int e = 0; e < 4; ++e) {
      unsigned u = __float_as_uint(v[e]);
      key[ch * 4 + e] = (u & 0x80000000u) ? ~u : (u | 0x80000000u);
    }
  }

  unsigned prefix = 0u;
  for (int bit = 31; bit >= 0; --bit) {
    const unsigned cand = prefix | (1u << bit);
    if (t == 0) s_cnt = 0;
    __syncthreads();
    int c = 0;
#pragma unroll
    for (int j = 0; j < 64; ++j) c += (key[j] >= cand) ? 1 : 0;
#pragma unroll
    for (int off = 16; off > 0; off >>= 1) c += __shfl_xor(c, off, 32);
    if ((t & 31) == 0) atomicAdd(&s_cnt, c);
    __syncthreads();
    if (s_cnt >= TOPK) prefix = cand;   // keep: still >= K keys above
    __syncthreads();
  }
  const unsigned thr = prefix;          // largest key with count(>=key) >= K

  if (t == 0) s_sel = 0;
  __syncthreads();
#pragma unroll
  for (int ch = 0; ch < 16; ++ch) {
    v4f out;
#pragma unroll
    for (int e = 0; e < 4; ++e) {
      const unsigned k = key[ch * 4 + e];
      const unsigned u = (k & 0x80000000u) ? (k ^ 0x80000000u) : ~k;
      const float v = __uint_as_float(u);
      const bool sel = (k >= thr);
      out[e] = sel ? v : 0.0f;
      if (sel) {
        int slot = atomicAdd(&s_sel, 1);
        if (slot < TOPK) { sv[slot] = v; si[slot] = ch * 1024 + t * 4 + e; }
      }
    }
    *(v4f*)(Lrow + ch * 1024 + t * 4) = out;   // sparse row (final output)
  }
  __syncthreads();

  // deterministic ordering: rank the 32 winners by latent index
  if (t < TOPK) {
    const int   myi = si[t];
    const float myv = sv[t];
    int rank = 0;
#pragma unroll
    for (int j = 0; j < TOPK; ++j) rank += (si[j] < myi) ? 1 : 0;
    outv[(size_t)row * TOPK + rank] = myv;
    outi[(size_t)row * TOPK + rank] = myi;
  }
}

// ====================== sparse decode (gather-GEMM) ========================
__global__ __launch_bounds__(256) void sae_decode(
    const float* __restrict__ Dec,  // [16384, 2048] (fits in 192MB L2)
    const float* __restrict__ vals, // [M, 32]
    const int*   __restrict__ inds, // [M, 32]
    float* __restrict__ Out)        // [M, 2048]
{
  __shared__ float s_v[TOPK];
  __shared__ int   s_i[TOPK];
  const int row = blockIdx.x;
  const int t   = threadIdx.x;
  if (t < TOPK) {
    s_v[t] = vals[(size_t)row * TOPK + t];
    s_i[t] = inds[(size_t)row * TOPK + t];
  }
  __syncthreads();

  // thread t owns cols [8t, 8t+8): two b128 loads per gathered decoder row
  v4f a0 = {}, a1 = {};
#pragma unroll 4
  for (int j = 0; j < TOPK; ++j) {
    const float v = s_v[j];
    const float* drow = Dec + (size_t)s_i[j] * HIDDEN_DIM + t * 8;
    v4f d0 = *(const v4f*)(drow);
    v4f d1 = *(const v4f*)(drow + 4);
#pragma unroll
    for (int e = 0; e < 4; ++e) { a0[e] += v * d0[e]; a1[e] += v * d1[e]; }
  }
  float* orow = Out + (size_t)row * HIDDEN_DIM + t * 8;
  __builtin_nontemporal_store(a0, (v4f*)orow);
  __builtin_nontemporal_store(a1, (v4f*)(orow + 4));
}

// ================================ launch ===================================
extern "C" void kernel_launch(void* const* d_in, const int* in_sizes, int n_in,
                              void* d_out, int out_size, void* d_ws, size_t ws_size,
                              hipStream_t stream) {
  (void)n_in; (void)out_size; (void)ws_size;
  const float* x   = (const float*)d_in[0];
  const float* enc = (const float*)d_in[1];
  const float* dec = (const float*)d_in[2];

  const int Mrows = in_sizes[0] / INPUT_DIM;   // 8192

  float* recon  = (float*)d_out;                               // [M, 2048]
  float* sparse = (float*)d_out + (size_t)Mrows * HIDDEN_DIM;  // [M, 16384]

  float* wv = (float*)d_ws;                                    // [M, 32] vals
  int*   wi = (int*)((char*)d_ws + (size_t)Mrows * TOPK * sizeof(float));

  // 1) encode GEMM: latent written into d_out's sparse region (scratch reuse)
  dim3 g1(LATENT_DIM / 128, Mrows / 128);
  sae_encode_bf16x3<<<g1, 256, 0, stream>>>(x, enc, sparse);

  // 2) per-row top-32 threshold + in-place sparsification + ranked compaction
  sae_topk<<<Mrows, 256, 0, stream>>>(sparse, wv, wi);

  // 3) sparse decode via 32-row gather (decoder stays L2-resident)
  sae_decode<<<Mrows, 256, 0, stream>>>(dec, wv, wi, recon);
}
